// GRU_23407571764060
// MI455X (gfx1250) — matrix operand order repaired and checked by reading, not compiled, not run on hardware
//
#include <hip/hip_runtime.h>
#include <hip/hip_bf16.h>

typedef __attribute__((ext_vector_type(16))) _Float16 v16h;
typedef __attribute__((ext_vector_type(8)))  _Float16 v8h;
typedef __attribute__((ext_vector_type(8)))  float    v8f;
typedef int v4i __attribute__((vector_size(16)));

#define AS1 __attribute__((address_space(1)))
#define AS3 __attribute__((address_space(3)))

#define HID   128
#define LSEQ  128
#define BT    16

// LDS byte offsets (all 16B aligned). Total = exactly 320 KB.
#define OFF_WHH1 0         // 96 KB : 4 ktiles x 24 ntiles x 1 KB fragments
#define OFF_WIH2 98304     // 96 KB
#define OFF_WHH2 196608    // 96 KB
#define OFF_WIH1 294912    // 12 KB : 24 frags x 512 B (K=16: lanes<16 read K 0..7, lanes>=16 K 8..15)
#define OFF_HA1  307200    // 2 x (16 x 272 B) ping-pong f16 A-layout
#define OFF_HA2  315904    // 2 x (16 x 272 B)
#define OFF_XA   324608    // 2 x (16 x 80 B)
#define OFF_ZPAGE 327168   // 512 B always-zero page (K=16 B-fragment elements 8..15)
#define SMEM_BYTES 327680
#define HA_STRIDE_H 136    // halves (272 B)
#define HA_BUF_H    2176   // halves per ping-pong buffer (4352 B)
#define XA_STRIDE_H 40     // halves (80 B)
#define XA_BUF_H    640    // halves per ping-pong buffer (1280 B)

__device__ __forceinline__ v8f wmma_f16(v16h a, v16h b, v8f c) {
  return __builtin_amdgcn_wmma_f32_16x16x32_f16(false, a, false, b, (short)0, c,
                                                false, false);
}

// A operand (16x32 f16): lane<16 -> row=lane, K {kb..kb+7, kb+16..kb+23}
//                        lane>=16 -> row=lane-16, K {kb+8..kb+15, kb+24..kb+31}
__device__ __forceinline__ v16h load_a(const _Float16* base, int strideH, int kb,
                                       int lane) {
  int row = lane & 15;
  int ko  = (lane < 16) ? 0 : 8;
  const _Float16* p = base + row * strideH + kb + ko;
  union { v16h v; v8h h[2]; } u;
  u.h[0] = *(const v8h*)(p);
  u.h[1] = *(const v8h*)(p + 16);
  return u.v;
}

// B operand fragment: 1 KB = two 512B blocks; block h: lane reads dwords 4h..4h+3
// at h*512 + lane*16 (consecutive lanes 4 banks apart -> conflict-friendly)
__device__ __forceinline__ v16h load_b(const char* frag, int lane) {
  union { v16h v; v8h h[2]; } u;
  u.h[0] = *(const v8h*)(frag + lane * 16);
  u.h[1] = *(const v8h*)(frag + 512 + lane * 16);
  return u.v;
}

// K=16 B fragment (w_ih1): lanes 0..15 hold K 0..7, lanes >=16 hold K 8..15 in
// elements 0..7; elements 8..15 (K 16..31) read from an always-zero LDS page.
__device__ __forceinline__ v16h load_b_k16(const char* frag, const char* zpage,
                                           int lane) {
  const char* p0 = (lane < 16) ? (frag + lane * 16) : (frag + 256 + (lane - 16) * 16);
  const char* p1 = zpage;
  union { v16h v; v8h h[2]; } u;
  u.h[0] = *(const v8h*)p0;
  u.h[1] = *(const v8h*)p1;
  return u.v;
}

// Fast gate nonlinearities on the hw trans pipes (v_exp/v_rcp/v_tanh).
__device__ __forceinline__ float fast_sigmoid(float x) {
  float e = __builtin_amdgcn_exp2f(-1.442695041f * x);
  return __builtin_amdgcn_rcpf(1.0f + e);
}
__device__ __forceinline__ float fast_tanh(float x) {
#if __has_builtin(__builtin_amdgcn_tanhf)
  return __builtin_amdgcn_tanhf(x);
#else
  float e = __builtin_amdgcn_exp2f(-2.885390082f * x);  // exp(-2x)
  return (1.0f - e) * __builtin_amdgcn_rcpf(1.0f + e);
#endif
}

// Pre-swizzle fp32 weights into the CDNA5 WMMA B-operand f16 layout.
// PW dword map: [WHH1: 24576][WIH2: 24576][WHH2: 24576][WIH1: 3072]
__global__ void pack_weights(const float* __restrict__ WIH1,
                             const float* __restrict__ WHH1,
                             const float* __restrict__ WIH2,
                             const float* __restrict__ WHH2,
                             unsigned int* __restrict__ PW) {
  int idx = blockIdx.x * 256 + threadIdx.x;
  if (idx >= 76800) return;
  const float* W; int K; int d;
  if (idx < 24576)      { W = WHH1; K = 128; d = idx; }
  else if (idx < 49152) { W = WIH2; K = 128; d = idx - 24576; }
  else if (idx < 73728) { W = WHH2; K = 128; d = idx - 49152; }
  else                  { W = WIH1; K = 16;  d = idx - 73728; }

  int n, k0;
  if (K == 128) {
    int f  = d >> 8;          // 256-dword fragment
    int r  = d & 255;
    int kt = f / 24, nt = f % 24;
    int h  = r >> 7;
    int rr = r & 127;
    int lane = rr >> 2;
    int v  = (h << 2) | (rr & 3);
    n  = nt * 16 + (lane & 15);                       // gate-output column
    k0 = kt * 32 + ((lane < 16) ? 0 : 8) + (h << 4) + 2 * (v & 3);    // K index (B[k][n]=W[n][k])
  } else {
    int f  = d >> 7;          // 128-dword fragment (16 lanes)
    int r  = d & 127;
    int h  = r >> 6;
    int rr = r & 63;
    int lane = rr >> 2;
    int v  = (h << 2) | (rr & 3);
    n  = f * 16 + lane;
    k0 = 2 * v;
  }
  _Float16 a = (_Float16)W[n * K + k0];
  _Float16 b = (_Float16)W[n * K + k0 + 1];
  union { _Float16 f; unsigned short u; } ca, cb;
  ca.f = a; cb.f = b;
  PW[idx] = (unsigned int)ca.u | ((unsigned int)cb.u << 16);
}

// Fully fused 2-layer GRU + FC. One workgroup = 16 batch rows, 8 waves;
// wave w owns hidden columns [16w, 16w+16) of both layers for all 128 steps.
// h/x tiles are ping-pong buffered by timestep parity -> 2 barriers per step.
__global__ void __launch_bounds__(256, 1)
gru_fused(const float* __restrict__ X, const unsigned int* __restrict__ PW,
          const float* __restrict__ BIH1, const float* __restrict__ BHH1,
          const float* __restrict__ BIH2, const float* __restrict__ BHH2,
          const float* __restrict__ FCW, const float* __restrict__ FCB,
          float* __restrict__ OUT) {
  extern __shared__ char smem[];
  const int tid  = threadIdx.x;
  const int lane = tid & 31;
  const int w    = tid >> 5;
  const int b0   = blockIdx.x * BT;

  // Zero h/x staging + zero page (1280 int4 = 20 KB tail of LDS).
  {
    int4* z = (int4*)(smem + OFF_HA1);
    int4 zero4 = {0, 0, 0, 0};
    for (int i = tid; i < 1280; i += 256) z[i] = zero4;
  }
  // Stage packed weights (300 KB) into LDS via the async-DMA path if available.
#if __has_builtin(__builtin_amdgcn_global_load_async_to_lds_b128) && \
    __has_builtin(__builtin_amdgcn_s_wait_asynccnt)
  {
    AS1 v4i* gsrc = (AS1 v4i*)(unsigned long long)PW;
    AS3 v4i* ldst = (AS3 v4i*)(unsigned)(unsigned long long)smem;
    for (int i = tid; i < 19200; i += 256)
      __builtin_amdgcn_global_load_async_to_lds_b128(gsrc + i, ldst + i, 0, 0);
    __builtin_amdgcn_s_wait_asynccnt(0);
  }
#else
  {
    const int4* src = (const int4*)PW;
    int4* dst = (int4*)smem;
    for (int i = tid; i < 19200; i += 256) dst[i] = src[i];
  }
#endif
  __syncthreads();

  _Float16* hA1 = (_Float16*)(smem + OFF_HA1);
  _Float16* hA2 = (_Float16*)(smem + OFF_HA2);
  _Float16* xA  = (_Float16*)(smem + OFF_XA);
  const char* WHH1 = smem + OFF_WHH1;
  const char* WIH2 = smem + OFF_WIH2;
  const char* WHH2 = smem + OFF_WHH2;
  const char* WIH1 = smem + OFF_WIH1;
  const char* ZPG  = smem + OFF_ZPAGE;

  // Per-lane gate biases (column = this lane's D-matrix N index).
  const int jcol = w * 16 + (lane & 15);
  const float b1r  = BIH1[jcol] + BHH1[jcol];
  const float b1z  = BIH1[HID + jcol] + BHH1[HID + jcol];
  const float b1in = BIH1[2 * HID + jcol];
  const float b1hn = BHH1[2 * HID + jcol];
  const float b2r  = BIH2[jcol] + BHH2[jcol];
  const float b2z  = BIH2[HID + jcol] + BHH2[HID + jcol];
  const float b2in = BIH2[2 * HID + jcol];
  const float b2hn = BHH2[2 * HID + jcol];

  // x reshape: (B,2,T)->(B,L,16): col c<8 from channel 0, c>=8 from channel 1.
  const int xrow = tid >> 4;
  const int xc   = tid & 15;
  const float* xptr =
      X + (size_t)(b0 + xrow) * 2048 + ((xc >> 3) << 10) + (xc & 7);

  // Prologue: stage x_0 into buffer 0 (visible after loop-top barrier of t=0).
  xA[xrow * XA_STRIDE_H + xc] = (_Float16)xptr[0];

  float h1p[8], h2p[8];
#pragma unroll
  for (int g = 0; g < 8; ++g) { h1p[g] = 0.f; h2p[g] = 0.f; }
  const int Mbase = (lane < 16) ? 0 : 8;   // D-matrix row base for this lane

#pragma unroll 1
  for (int t = 0; t < LSEQ; ++t) {
    const int p = t & 1;
    const _Float16* hA1r = hA1 + p * HA_BUF_H;
    _Float16*       hA1w = hA1 + (1 - p) * HA_BUF_H;
    const _Float16* hA2r = hA2 + p * HA_BUF_H;
    _Float16*       hA2w = hA2 + (1 - p) * HA_BUF_H;
    const _Float16* xAr  = xA + p * XA_BUF_H;
    _Float16*       xAw  = xA + (1 - p) * XA_BUF_H;

    __syncthreads();   // barrier 1: x_t, h1(t-1), h2(t-1) all visible

    // Stage x for t+1 into the alternate buffer (consumed next iteration).
    const int tn = (t + 1) & (LSEQ - 1);
    xAw[xrow * XA_STRIDE_H + xc] = (_Float16)xptr[tn * 8];
    __builtin_prefetch(xptr + (((t + 2) & (LSEQ - 1)) * 8), 0, 0);

    // ================= layer 1 =================
    v16h ax = load_a(xAr, XA_STRIDE_H, 0, lane);
    v16h a1[4];
#pragma unroll
    for (int kt = 0; kt < 4; ++kt) a1[kt] = load_a(hA1r, HA_STRIDE_H, kt * 32, lane);

    v8f accR = {}, accZ = {}, accNi = {}, accNh = {};
    accR  = wmma_f16(ax, load_b_k16(WIH1 + (size_t)w * 512, ZPG, lane), accR);
    accZ  = wmma_f16(ax, load_b_k16(WIH1 + (size_t)(8 + w) * 512, ZPG, lane), accZ);
    accNi = wmma_f16(ax, load_b_k16(WIH1 + (size_t)(16 + w) * 512, ZPG, lane), accNi);
#pragma unroll
    for (int kt = 0; kt < 4; ++kt) {
      accR  = wmma_f16(a1[kt], load_b(WHH1 + (size_t)(kt * 24 + w) * 1024, lane), accR);
      accZ  = wmma_f16(a1[kt], load_b(WHH1 + (size_t)(kt * 24 + 8 + w) * 1024, lane), accZ);
      accNh = wmma_f16(a1[kt], load_b(WHH1 + (size_t)(kt * 24 + 16 + w) * 1024, lane), accNh);
    }

    float h1n[8];
#pragma unroll
    for (int g = 0; g < 8; ++g) {
      float r = fast_sigmoid(accR[g] + b1r);
      float z = fast_sigmoid(accZ[g] + b1z);
      float n = fast_tanh(accNi[g] + b1in + r * (accNh[g] + b1hn));
      h1n[g] = n + z * (h1p[g] - n);
      h1p[g] = h1n[g];
    }
    // Write h1(t) into the write buffer (nobody reads it until barrier 2).
#pragma unroll
    for (int g = 0; g < 8; ++g)
      hA1w[(Mbase + g) * HA_STRIDE_H + jcol] = (_Float16)h1n[g];

    __syncthreads();   // barrier 2: h1(t) visible for layer 2

    // ================= layer 2 =================
    v16h ai[4], ah[4];
#pragma unroll
    for (int kt = 0; kt < 4; ++kt) {
      ai[kt] = load_a(hA1w, HA_STRIDE_H, kt * 32, lane);
      ah[kt] = load_a(hA2r, HA_STRIDE_H, kt * 32, lane);
    }
    v8f cR = {}, cZ = {}, cNi = {}, cNh = {};
#pragma unroll
    for (int kt = 0; kt < 4; ++kt) {
      cR  = wmma_f16(ai[kt], load_b(WIH2 + (size_t)(kt * 24 + w) * 1024, lane), cR);
      cZ  = wmma_f16(ai[kt], load_b(WIH2 + (size_t)(kt * 24 + 8 + w) * 1024, lane), cZ);
      cNi = wmma_f16(ai[kt], load_b(WIH2 + (size_t)(kt * 24 + 16 + w) * 1024, lane), cNi);
      cR  = wmma_f16(ah[kt], load_b(WHH2 + (size_t)(kt * 24 + w) * 1024, lane), cR);
      cZ  = wmma_f16(ah[kt], load_b(WHH2 + (size_t)(kt * 24 + 8 + w) * 1024, lane), cZ);
      cNh = wmma_f16(ah[kt], load_b(WHH2 + (size_t)(kt * 24 + 16 + w) * 1024, lane), cNh);
    }

    float h2n[8];
#pragma unroll
    for (int g = 0; g < 8; ++g) {
      float r = fast_sigmoid(cR[g] + b2r);
      float z = fast_sigmoid(cZ[g] + b2z);
      float n = fast_tanh(cNi[g] + b2in + r * (cNh[g] + b2hn));
      h2n[g] = n + z * (h2p[g] - n);
      h2p[g] = h2n[g];
    }
    // Write h2(t); next loop-top barrier publishes it.
#pragma unroll
    for (int g = 0; g < 8; ++g)
      hA2w[(Mbase + g) * HA_STRIDE_H + jcol] = (_Float16)h2n[g];
  }

  __syncthreads();
  // FC head: publish final h2 (fp32) into LDS (reusing WIH1 region), then
  // 16x24 dot products of length 128 in fp32.
  float* H2 = (float*)(smem + OFF_WIH1);
#pragma unroll
  for (int g = 0; g < 8; ++g) H2[(Mbase + g) * HID + jcol] = h2p[g];
  __syncthreads();

  for (int idx = tid; idx < BT * 24; idx += 256) {
    int row = idx / 24;
    int cls = idx - row * 24;
    float s = FCB[cls];
    const float* hrow = H2 + row * HID;
    const float* wrow = FCW + cls * HID;
#pragma unroll 8
    for (int k = 0; k < HID; ++k) s += hrow[k] * wrow[k];
    OUT[(size_t)(b0 + row) * 24 + cls] = s;
  }
}

extern "C" void kernel_launch(void* const* d_in, const int* in_sizes, int n_in,
                              void* d_out, int out_size, void* d_ws,
                              size_t ws_size, hipStream_t stream) {
  const float* x    = (const float*)d_in[0];
  const float* wih1 = (const float*)d_in[1];
  const float* whh1 = (const float*)d_in[2];
  const float* bih1 = (const float*)d_in[3];
  const float* bhh1 = (const float*)d_in[4];
  const float* wih2 = (const float*)d_in[5];
  const float* whh2 = (const float*)d_in[6];
  const float* bih2 = (const float*)d_in[7];
  const float* bhh2 = (const float*)d_in[8];
  const float* fcw  = (const float*)d_in[9];
  const float* fcb  = (const float*)d_in[10];
  float* out = (float*)d_out;
  unsigned int* pw = (unsigned int*)d_ws;   // 307200 bytes of packed f16 weights

  pack_weights<<<300, 256, 0, stream>>>(wih1, whh1, wih2, whh2, pw);
  gru_fused<<<4096 / BT, 256, SMEM_BYTES, stream>>>(x, pw, bih1, bhh1, bih2,
                                                    bhh2, fcw, fcb, out);
}
